// Resonator_67705864454650
// MI455X (gfx1250) — 8-bit matrix operand order repaired and checked by reading, compiled, not run on hardware
//
#include <hip/hip_runtime.h>
#include <hip/hip_bf16.h>
#include <stdint.h>


#define F 4
#define M 256
#define D 8192
#define B 256
#define ITERS 10

typedef __attribute__((ext_vector_type(8))) int v8i;


__global__ void k_flags(int* flags) {
    int t = threadIdx.x;
    if (t < ITERS) flags[t] = (t == 0) ? 1 : 0;
}

__global__ void k_quant_cb(const float* __restrict__ cb,
                           int8_t* __restrict__ cb8,
                           int8_t* __restrict__ cbt8) {
    size_t i = (size_t)blockIdx.x * blockDim.x + threadIdx.x;
    if (i >= (size_t)F * M * D) return;
    int d = (int)(i % D);
    size_t fm = i / D;
    int m = (int)(fm % M);
    int f = (int)(fm / M);
    float v = cb[i];
    int8_t q = (int8_t)((v > 0.f) - (v < 0.f));
    cb8[i] = q;
    cbt8[((size_t)f * D + d) * M + m] = q;
}

__global__ void k_quant_in(const float* __restrict__ in, int8_t* __restrict__ in8) {
    size_t i = (size_t)blockIdx.x * blockDim.x + threadIdx.x;
    if (i >= (size_t)B * D) return;
    float v = in[i];
    in8[i] = (int8_t)((v > 0.f) - (v < 0.f));
}

__global__ void k_colsum(const float* __restrict__ cb, float* __restrict__ cs) {
    int i = blockIdx.x * blockDim.x + threadIdx.x;
    if (i >= F * D) return;
    int f = i / D, d = i % D;
    float s = 0.f;
    for (int m = 0; m < M; ++m) s += cb[((size_t)f * M + m) * D + d];
    cs[i] = s;
}

__global__ void k_inv0(const float* __restrict__ cs, float* __restrict__ inv0) {
    int i = blockIdx.x * blockDim.x + threadIdx.x;
    if (i >= F * D) return;
    int f = i / D, d = i % D;
    float p = 1.f;
    for (int ff = 0; ff < F; ++ff)
        if (ff != f) p *= cs[ff * D + d];
    inv0[i] = p;
}

__global__ void k_sim0(const float* __restrict__ input, const float* __restrict__ inv0,
                       const int8_t* __restrict__ cbt8, float* __restrict__ sim0) {
    __shared__ float v[D];
    int g = blockIdx.x;
    int b = g & (B - 1);
    int f = g >> 8;
    for (int d = threadIdx.x; d < D; d += 256)
        v[d] = input[(size_t)b * D + d] * inv0[f * D + d];
    __syncthreads();
    int m = threadIdx.x;
    const int8_t* cp = cbt8 + (size_t)f * D * M + m;
    float acc = 0.f;
    for (int d = 0; d < D; ++d)
        acc += v[d] * (float)cp[(size_t)d * M];
    sim0[(size_t)g * M + m] = acc;
}

__global__ void k_out0(const float* __restrict__ sim0, const int8_t* __restrict__ cb8,
                       int8_t* __restrict__ e1) {
    __shared__ float sr[M];
    int g  = blockIdx.x;
    int dc = g & 31;
    int fb = g >> 5;
    int f  = fb >> 8;
    sr[threadIdx.x] = sim0[(size_t)fb * M + threadIdx.x];
    __syncthreads();
    int d = dc * 256 + threadIdx.x;
    const int8_t* cp = cb8 + (size_t)f * M * D + d;
    float acc = 0.f;
    for (int m = 0; m < M; ++m) acc += sr[m] * (float)cp[(size_t)m * D];
    e1[(size_t)fb * D + d] = (int8_t)((acc > 0.f) - (acc < 0.f));
}


__global__ void k_buildA(const int8_t* __restrict__ in8, const int8_t* __restrict__ es,
                         int8_t* __restrict__ a8) {
    size_t i = (size_t)blockIdx.x * blockDim.x + threadIdx.x;
    if (i >= (size_t)B * D) return;
    const size_t BD = (size_t)B * D;
    int x  = in8[i];
    int e0 = es[i], e1 = es[BD + i], e2 = es[2 * BD + i], e3 = es[3 * BD + i];
    a8[i]          = (int8_t)(x * e1 * e2 * e3);
    a8[BD + i]     = (int8_t)(x * e0 * e2 * e3);
    a8[2 * BD + i] = (int8_t)(x * e0 * e1 * e3);
    a8[3 * BD + i] = (int8_t)(x * e0 * e1 * e2);
}

__global__ void k_gemm1(const int8_t* __restrict__ a8, const int8_t* __restrict__ cb8,
                        int* __restrict__ sim) {
    int wave = blockIdx.x * (blockDim.x >> 5) + (threadIdx.x >> 5);
    int mt   = wave & 15;
    int bt   = (wave >> 4) & 15;
    int f    = wave >> 8;
    int lane = threadIdx.x & 31;
    int l15  = lane & 15;
    int half = lane >> 4;

    const int8_t* ap = a8  + ((size_t)f * B + bt * 16 + l15) * D + half * 8;
    const int8_t* bp = cb8 + ((size_t)f * M + mt * 16 + l15) * D + half * 8;

    v8i acc; for (int r = 0; r < 8; ++r) acc[r] = 0;
    for (int k = 0; k < D; k += 64) {
        const int* ai = (const int*)(ap + k);
        const int* bi = (const int*)(bp + k);
        v8i a, b;
        a[0] = ai[0];  a[1] = ai[1];  a[2] = ai[4];  a[3] = ai[5];
        a[4] = ai[8];  a[5] = ai[9];  a[6] = ai[12]; a[7] = ai[13];
        b[0] = bi[0];  b[1] = bi[1];  b[2] = bi[4];  b[3] = bi[5];
        b[4] = bi[8];  b[5] = bi[9];  b[6] = bi[12]; b[7] = bi[13];
        acc = __builtin_amdgcn_wmma_i32_16x16x64_iu8(true, a, true, b, acc, false, false);
    }
    int colm = mt * 16 + l15;
    int rowb = bt * 16 + half * 8;
    for (int r = 0; r < 8; ++r)
        sim[((size_t)f * B + rowb + r) * M + colm] = acc[r];
}

__global__ void k_gemm2(const int* __restrict__ sim, const int8_t* __restrict__ cbt8,
                        const int8_t* __restrict__ esrc, int8_t* __restrict__ edst,
                        int* __restrict__ changed, int iter) {
    int wave = blockIdx.x * (blockDim.x >> 5) + (threadIdx.x >> 5);
    int dt   = wave & 511;
    int bt   = (wave >> 9) & 15;
    int f    = wave >> 13;
    int lane = threadIdx.x & 31;
    int l15  = lane & 15;
    int half = lane >> 4;

    const int* srow = sim + ((size_t)f * B + bt * 16 + l15) * M;
    const int8_t* bp = cbt8 + ((size_t)f * D + dt * 16 + l15) * M + half * 8;

    v8i chi, clo;
    for (int r = 0; r < 8; ++r) { chi[r] = 0; clo[r] = 0; }

    for (int kk = 0; kk < 4; ++kk) {
        int kbase = kk * 64 + half * 8;
        v8i ahi, alo;
        for (int g = 0; g < 4; ++g) {
            int m0 = kbase + g * 16;
            int w0h = 0, w1h = 0, w0l = 0, w1l = 0;
            for (int j = 0; j < 8; ++j) {
                int s  = srow[m0 + j];
                int hi = (s + 64) >> 7;
                int lo = s - (hi << 7);
                if (j < 4) { w0h |= (hi & 255) << (8 * j);       w0l |= (lo & 255) << (8 * j); }
                else       { w1h |= (hi & 255) << (8 * (j - 4)); w1l |= (lo & 255) << (8 * (j - 4)); }
            }
            ahi[2 * g] = w0h; ahi[2 * g + 1] = w1h;
            alo[2 * g] = w0l; alo[2 * g + 1] = w1l;
        }
        const int* bi = (const int*)(bp + kk * 64);
        v8i b;
        b[0] = bi[0]; b[1] = bi[1]; b[2] = bi[4];  b[3] = bi[5];
        b[4] = bi[8]; b[5] = bi[9]; b[6] = bi[12]; b[7] = bi[13];
        chi = __builtin_amdgcn_wmma_i32_16x16x64_iu8(true, ahi, true, b, chi, false, false);
        clo = __builtin_amdgcn_wmma_i32_16x16x64_iu8(true, alo, true, b, clo, false, false);
    }

    int dcol = dt * 16 + l15;
    int rowb = bt * 16 + half * 8;
    int diff = 0;
    for (int r = 0; r < 8; ++r) {
        int o = chi[r] * 128 + clo[r];
        int8_t s = (int8_t)((o > 0) - (o < 0));
        size_t idx = ((size_t)f * B + rowb + r) * D + dcol;
        diff |= (s != esrc[idx]);
        edst[idx] = s;
    }
    if (diff && changed[iter] == 0) atomicOr(&changed[iter], 1);
}


__global__ void k_argmax(const int* __restrict__ sim, float* __restrict__ out) {
    int t = blockIdx.x * blockDim.x + threadIdx.x;
    if (t >= B * F) return;
    int b = t >> 2, f = t & 3;
    const int* s = sim + ((size_t)f * B + b) * M;
    int best = s[0], bi = 0;
    for (int m = 1; m < M; ++m)
        if (s[m] > best) { best = s[m]; bi = m; }
    out[t] = (float)bi;
}

__global__ void k_fink(const int* __restrict__ changed, float* __restrict__ out) {
    if (threadIdx.x == 0 && blockIdx.x == 0) {
        int k = ITERS;
        for (int j = 0; j < ITERS; ++j)
            if (changed[j] == 0) { k = j + 1; break; }
        out[B * F] = (float)k;
    }
}


extern "C" void kernel_launch(void* const* d_in, const int* in_sizes, int n_in,
                              void* d_out, int out_size, void* d_ws, size_t ws_size,
                              hipStream_t stream) {
    (void)in_sizes; (void)n_in; (void)out_size; (void)ws_size;

    const float* input = (const float*)d_in[0];
    const float* cb    = (const float*)d_in[1];
    float* out = (float*)d_out;

    uint8_t* w = (uint8_t*)d_ws;
    const size_t SZ1 = (size_t)F * M * D;
    int8_t* cb8   = (int8_t*)(w);
    int8_t* cbt8  = (int8_t*)(w + SZ1);
    int8_t* in8   = (int8_t*)(w + 2 * SZ1);
    int8_t* a8    = (int8_t*)(w + 2 * SZ1 + (size_t)B * D);
    int8_t* e0    = (int8_t*)(w + 3 * SZ1 + (size_t)B * D);
    int8_t* e1    = (int8_t*)(w + 4 * SZ1 + (size_t)B * D);
    int*    sim   = (int*)  (w + 5 * SZ1 + (size_t)B * D);
    float*  sim0  = (float*)sim;
    float*  cs    = (float*)(w + 5 * SZ1 + (size_t)B * D + (size_t)F * B * M * 4);
    float*  inv0  = cs + F * D;
    int*    flags = (int*)(inv0 + F * D);

    k_flags   <<<1, 32, 0, stream>>>(flags);
    k_quant_cb<<<(int)(SZ1 / 256), 256, 0, stream>>>(cb, cb8, cbt8);
    k_quant_in<<<(B * D) / 256, 256, 0, stream>>>(input, in8);
    k_colsum  <<<(F * D) / 256, 256, 0, stream>>>(cb, cs);
    k_inv0    <<<(F * D) / 256, 256, 0, stream>>>(cs, inv0);

    k_sim0<<<B * F, 256, 0, stream>>>(input, inv0, cbt8, sim0);
    k_out0<<<F * B * (D / 256), 256, 0, stream>>>(sim0, cb8, e1);

    for (int j = 1; j < ITERS; ++j) {
        int8_t* src = (j & 1) ? e1 : e0;
        int8_t* dst = (j & 1) ? e0 : e1;
        k_buildA<<<(B * D) / 256, 256, 0, stream>>>(in8, src, a8);
        k_gemm1 <<<(F * 16 * 16) / 8, 256, 0, stream>>>(a8, cb8, sim);
        k_gemm2 <<<(F * 16 * (D / 16)) / 8, 256, 0, stream>>>(sim, cbt8, src, dst, flags, j);
    }

    k_gemm1 <<<(F * 16 * 16) / 8, 256, 0, stream>>>(e0, cb8, sim);
    k_argmax<<<(B * F) / 256, 256, 0, stream>>>(sim, out);
    k_fink  <<<1, 1, 0, stream>>>(flags, out);
}
